// R3D2Net_30975304139468
// MI455X (gfx1250) — matrix operand order repaired and checked by reading, not compiled, not run on hardware
//
#include <hip/hip_runtime.h>
#include <hip/hip_bf16.h>
#include <math.h>

// ---------------------------------------------------------------------------
// R3D2Net forward on gfx1250:
//   relu-GEMM -> hoisted x@Wi GEMM -> persistent-kernel 128-step LSTM scan
//   (Wh staged in LDS, c in registers, software-pipelined fragments,
//    branchless nonlinearities, Gx preloaded behind the grid barrier)
//   -> proj GEMM -> action/value heads.
// All matrix math: v_wmma_f32_16x16x32_bf16 (bf16 operands, f32 accum).
// ---------------------------------------------------------------------------

#define T_DIM 128
#define B_DIM 64
#define D_STATE 512
#define H_DIM 512
#define G_DIM 2048       // 4*H
#define D_BERT 768
#define A_DIM 20
#define ROWS (T_DIM * B_DIM)   // 8192
#define SCAN_BLOCKS (H_DIM / 16)  // 32

typedef __attribute__((ext_vector_type(16))) __bf16        v16bf;
typedef __attribute__((ext_vector_type(8)))  float         v8f;
typedef __attribute__((ext_vector_type(8)))  unsigned int  v8u;

union frag_cast { v16bf bf; v8u u; };

static __device__ inline v8f wmma_bf16(v16bf a, v16bf b, v8f c) {
  // (neg_a, A, neg_b, B, c_mod, C, reuse_a, reuse_b)
  return __builtin_amdgcn_wmma_f32_16x16x32_bf16(false, a, false, b, (short)0, c,
                                                 false, false);
}

// branchless sigmoid / tanh (v_exp_f32 based; no EXEC-mask branches)
static __device__ inline float sigm(float x) { return 1.0f / (1.0f + __expf(-x)); }
static __device__ inline float tanh_fast(float x) {
  return 1.0f - 2.0f / (__expf(2.0f * x) + 1.0f);  // saturates to +-1 correctly
}

// ===========================================================================
// Fragment layouts (CDNA5 ISA 7.12.2, wave32):
//  A (16x32 bf16): lane L holds row M=L%16; K values {kh..kh+7, kh+16..kh+23}
//                  with kh = 8*(L/16); element e -> K = kh + (e&7) + ((e>>3)<<4).
//  B (32x16 bf16): lane L holds col N=L%16; same K values as A, kh = 8*(L/16);
//                  element e -> K = kh + (e&7) + ((e>>3)<<4).
//  C/D (16x16 f32): VGPR r -> row r + 8*(L/16), col L%16.
// ===========================================================================

// --- packed-B loader: fragment-major, one 32B vector load per lane ----------
static __device__ inline v16bf load_b_packed(const __bf16* __restrict__ Bp,
                                             int kt, int nt, int ntiles) {
  int lane = threadIdx.x & 31;
  return ((const v16bf*)Bp)[(size_t)(kt * ntiles + nt) * 32 + lane];
}

// --- A fragment from row-major bf16 activations (two 16B loads) -------------
static __device__ inline v16bf load_a_frag(const __bf16* __restrict__ A,
                                           int row_base, int lda, int kb) {
  int lane = threadIdx.x & 31;
  int m = row_base + (lane & 15);
  int khalf = (lane >> 4) * 8;
  const __bf16* p = A + (size_t)m * lda + kb + khalf;
  v16bf a;
#pragma unroll
  for (int e = 0; e < 8; ++e) {
    a[e]     = p[e];        // K = khalf + e        (16B contiguous)
    a[8 + e] = p[16 + e];   // K = khalf + 16 + e   (16B contiguous)
  }
  return a;
}

// --- A fragment from packed hidden state + precomputed done mask ------------
// Mask applied as 8x v_cndmask_b32 (SGPR lane-mask cond; EXEC unchanged).
static __device__ inline v16bf load_a_hpack(const __bf16* __restrict__ hp,
                                            bool dz, int wave, int kt) {
  int lane = threadIdx.x & 31;
  frag_cast f;
  f.u = ((const v8u*)hp)[(size_t)(kt * 4 + wave) * 32 + lane];
#pragma unroll
  for (int e = 0; e < 8; ++e) f.u[e] = dz ? 0u : f.u[e];
  return f.bf;
}

// ===========================================================================
// Weight pre-swizzle: f32 row-major [K,N] -> bf16 fragment-major packed.
// ===========================================================================
__global__ void pack_b_kernel(const float* __restrict__ W, __bf16* __restrict__ out,
                              int K, int N) {
  int idx = blockIdx.x * blockDim.x + threadIdx.x;
  if (idx >= K * N) return;
  int e    = idx & 15;
  int lane = (idx >> 4) & 31;
  int tile = idx >> 9;
  int ntiles = N >> 4;
  int nt = tile % ntiles;
  int kt = tile / ntiles;
  int k = kt * 32 + ((lane >> 4) << 3) + (e & 7) + ((e >> 3) << 4);
  int n = nt * 16 + (lane & 15);
  out[idx] = (__bf16)W[(size_t)k * N + n];
}

// ===========================================================================
// Generic WMMA GEMM: C[M,N] = op(A[M,K] @ Bpacked[K,N] + bias)
// Block = 128 threads (4 waves), block tile 64x64, wave reuses its A fragment
// across 4 N-tiles. grid = (N/64, M/64). Latency hidden by multi-wave occupancy.
// ===========================================================================
template <bool RELU, bool OUT_BF16>
__global__ void gemm_wmma_kernel(const __bf16* __restrict__ A,
                                 const __bf16* __restrict__ Bp,
                                 const float* __restrict__ bias,
                                 void* __restrict__ Cout, int K, int N) {
  int wave = threadIdx.x >> 5;
  int lane = threadIdx.x & 31;
  int row_base = blockIdx.y * 64 + wave * 16;
  int nt0 = blockIdx.x * 4;
  int ntiles = N >> 4;
  v8f acc[4] = {};
#pragma unroll 2
  for (int kt = 0; kt < (K >> 5); ++kt) {
    v16bf a = load_a_frag(A, row_base, K, kt * 32);
#pragma unroll
    for (int j = 0; j < 4; ++j) {
      v16bf b = load_b_packed(Bp, kt, nt0 + j, ntiles);
      acc[j] = wmma_bf16(a, b, acc[j]);
    }
  }
  int n = lane & 15;
  int mo = (lane >> 4) * 8;
#pragma unroll
  for (int j = 0; j < 4; ++j) {
    int col = (nt0 + j) * 16 + n;
    float bv = bias[col];
#pragma unroll
    for (int r = 0; r < 8; ++r) {
      int row = row_base + mo + r;
      float val = acc[j][r] + bv;
      if (RELU) val = fmaxf(val, 0.0f);
      size_t idx = (size_t)row * N + col;
      if (OUT_BF16) ((__bf16*)Cout)[idx] = (__bf16)val;
      else          ((float*)Cout)[idx] = val;
    }
  }
}

// ===========================================================================
// Device-scope grid barrier for the persistent scan kernel.
// Monotonic counter (zeroed per call) -> deterministic, graph-replay safe.
// ===========================================================================
static __device__ inline void grid_barrier(unsigned int* counter, unsigned int target) {
  __syncthreads();
  if (threadIdx.x == 0) {
    __hip_atomic_fetch_add(counter, 1u, __ATOMIC_RELEASE, __HIP_MEMORY_SCOPE_AGENT);
    while (__hip_atomic_load(counter, __ATOMIC_ACQUIRE, __HIP_MEMORY_SCOPE_AGENT) < target) {
      __builtin_amdgcn_s_sleep(1);
    }
  }
  __syncthreads();
}

// ===========================================================================
// Persistent LSTM scan: ONE kernel runs all 128 timesteps.
// grid = 32 blocks (one 16-wide H column group across all 4 gates each),
// block = 128 threads (4 waves, one 16-row batch strip each).
//  - Wh working set (64KB, step-invariant) staged into LDS once.
//  - Cell state c lives in 8 VGPRs/thread for the whole scan.
//  - A/B fragments explicitly double-buffered: fragment f+1's loads issue
//    before the WMMA consuming fragment f (hides ds/global latency at
//    single-wave occupancy).
//  - Next step's Gx/done preloaded into registers BEFORE the grid barrier.
// Final step writes cT|hT directly into d_out.
// ===========================================================================
__global__ void lstm_scan_kernel(const __bf16* __restrict__ Whp,   // packed B
                                 const float* __restrict__ Gx,     // [T,64,2048], has b_lstm
                                 const unsigned char* __restrict__ dones,  // [T,64]
                                 const float* __restrict__ hidden_c,       // [64,512] input
                                 __bf16* __restrict__ hp0,         // packed A ping
                                 __bf16* __restrict__ hp1,         // packed A pong
                                 __bf16* __restrict__ ys,          // [T,64,512] bf16
                                 float* __restrict__ out_ch,       // d_out: cT | hT
                                 unsigned int* __restrict__ counter) {
  __shared__ v16bf whl[64 * 32];       // 64 fragments x 32 lanes x 32B = 64KB

  int wave = threadIdx.x >> 5;
  int lane = threadIdx.x & 31;
  int ncol = blockIdx.x * 16;
  int n = lane & 15;
  int mo = (lane >> 4) * 8;
  int kt_w = ncol >> 5;                // K-tile (next step) this column group feeds
  int kl = (ncol & 31) + n;

  // ---- stage this block's Wh fragments (step-invariant) into LDS ----------
  for (int i = threadIdx.x; i < 64 * 32; i += 128) {
    int f = i >> 5;                    // fragment: kt*4 + g
    int l = i & 31;
    int kt = f >> 2, g = f & 3;
    whl[i] = ((const v16bf*)Whp)[((size_t)kt * (G_DIM >> 4) +
                                  ((g * H_DIM + ncol) >> 4)) * 32 + l];
  }
  __syncthreads();

  // ---- cell state in registers for the whole scan -------------------------
  float c_reg[8];
#pragma unroll
  for (int r = 0; r < 8; ++r)
    c_reg[r] = hidden_c[(size_t)((wave * 16) + mo + r) * H_DIM + ncol + n];

  // ---- preload step 0 gate pre-activations + done flags -------------------
  float gx[4][8];
  bool dz_a;
  unsigned dmask;
  auto preload = [&](int t) {
    const float* Gt = Gx + (size_t)t * B_DIM * G_DIM;
    const unsigned char* dt = dones + (size_t)t * B_DIM;
#pragma unroll
    for (int r = 0; r < 8; ++r) {
      size_t gbase = (size_t)((wave * 16) + mo + r) * G_DIM + ncol + n;
#pragma unroll
      for (int g = 0; g < 4; ++g) gx[g][r] = Gt[gbase + g * H_DIM];
    }
    dz_a = dt[wave * 16 + (lane & 15)] != 0;
    dmask = 0;
#pragma unroll
    for (int r = 0; r < 8; ++r) dmask |= (dt[(wave * 16) + mo + r] ? 1u : 0u) << r;
  };
  preload(0);

  for (int t = 0; t < T_DIM; ++t) {
    const __bf16* hp_in = (t & 1) ? hp1 : hp0;
    __bf16* hp_out      = (t & 1) ? hp0 : hp1;

    // ---- software-pipelined WMMA over 64 fragments (16 K-tiles x 4 gates) --
    v8f acc[4] = {};
    v16bf a_cur = load_a_hpack(hp_in, dz_a, wave, 0);
    v16bf b_cur = whl[lane];           // fragment f = 0
#pragma unroll
    for (int f = 0; f < 64; ++f) {
      const int g = f & 3;
      const int kt = f >> 2;
      v16bf b_nxt = (f < 63) ? whl[(f + 1) * 32 + lane] : b_cur;
      v16bf a_nxt = (g == 3 && kt < 15) ? load_a_hpack(hp_in, dz_a, wave, kt + 1)
                                        : a_cur;
      acc[g] = wmma_bf16(a_cur, b_cur, acc[g]);
      b_cur = b_nxt;
      a_cur = a_nxt;
    }

#pragma unroll
    for (int r = 0; r < 8; ++r) {
      int row = (wave * 16) + mo + r;              // batch index 0..63
      int hcol = ncol + n;
      float gi = acc[0][r] + gx[0][r];
      float gf = acc[1][r] + gx[1][r];
      float gg = acc[2][r] + gx[2][r];
      float go = acc[3][r] + gx[3][r];
      float cin = ((dmask >> r) & 1u) ? 0.0f : c_reg[r];
      float nc = sigm(gf) * cin + sigm(gi) * tanh_fast(gg);
      float nh = sigm(go) * tanh_fast(nc);
      c_reg[r] = nc;
      size_t hidx = (size_t)row * H_DIM + hcol;
      ys[(size_t)t * B_DIM * H_DIM + hidx] = (__bf16)nh;
      if (t == T_DIM - 1) {
        out_ch[hidx] = nc;                          // cT
        out_ch[(size_t)B_DIM * H_DIM + hidx] = nh;  // hT
      } else {
        // scatter into packed A-fragment layout for the next step
        int m_tile = row >> 4, m = row & 15;
        int plane = m + (((kl >> 3) & 1) << 4);
        int pe    = (kl & 7) + ((kl >> 4) << 3);
        hp_out[((size_t)(kt_w * 4 + m_tile) * 32 + plane) * 16 + pe] = (__bf16)nh;
      }
    }

    if (t != T_DIM - 1) {
      preload(t + 1);   // independent loads issued before the barrier: latency
                        // overlaps with the barrier wait
      grid_barrier(counter, (unsigned int)(SCAN_BLOCKS * (t + 1)));
    }
  }
}

// ===========================================================================
// Small helper kernels
// ===========================================================================
__global__ void cvt_f32_bf16_kernel(const float* __restrict__ in,
                                    __bf16* __restrict__ out, int n) {
  int i = blockIdx.x * blockDim.x + threadIdx.x;
  if (i < n) out[i] = (__bf16)in[i];
}

// init h (packed bf16) + barrier counter; i covers B*H
__global__ void init_h_kernel(const float* __restrict__ hh,
                              __bf16* __restrict__ hp,
                              unsigned int* __restrict__ counter, int n) {
  int i = blockIdx.x * blockDim.x + threadIdx.x;
  if (i == 0) *counter = 0u;
  if (i >= n) return;
  int row = i >> 9, k = i & (H_DIM - 1);
  int kt = k >> 5, kl = k & 31;
  int m_tile = row >> 4, m = row & 15;
  int plane = m + (((kl >> 3) & 1) << 4);
  int pe    = (kl & 7) + ((kl >> 4) << 3);
  hp[((size_t)(kt * 4 + m_tile) * 32 + plane) * 16 + pe] = (__bf16)hh[i];
}

__global__ void v_kernel(const __bf16* __restrict__ ys,
                         const float* __restrict__ Wv,
                         const float* __restrict__ bv,
                         float* __restrict__ v) {
  int r = blockIdx.x * blockDim.x + threadIdx.x;
  if (r >= ROWS) return;
  float s = bv[0];
  const v16bf* p = (const v16bf*)(ys + (size_t)r * H_DIM);
#pragma unroll 4
  for (int kb = 0; kb < H_DIM / 16; ++kb) {
    v16bf x = p[kb];
#pragma unroll
    for (int e = 0; e < 16; ++e) s += (float)x[e] * Wv[kb * 16 + e];
  }
  v[r] = s;
}

// aw[a][d] = action_emb[a][d] * Wa[d]
__global__ void aw_kernel(const float* __restrict__ action_emb,
                          const float* __restrict__ Wa,
                          float* __restrict__ aw) {
  int i = blockIdx.x * blockDim.x + threadIdx.x;
  if (i < A_DIM * D_BERT) aw[i] = action_emb[i] * Wa[i % D_BERT];
}

__global__ void q_kernel(const float* __restrict__ proj,
                         const float* __restrict__ aw,
                         const float* __restrict__ ba,
                         const float* __restrict__ v,
                         float* __restrict__ q) {
  int idx = blockIdx.x * blockDim.x + threadIdx.x;
  if (idx >= ROWS * A_DIM) return;
  int r = idx / A_DIM;
  int a = idx % A_DIM;
  const float* pp = proj + (size_t)r * D_BERT;
  const float* pa = aw + (size_t)a * D_BERT;
  float s = 0.0f;
#pragma unroll 4
  for (int d = 0; d < D_BERT; ++d) s += pp[d] * pa[d];
  q[idx] = v[r] + ba[0] + s;
}

// ===========================================================================
extern "C" void kernel_launch(void* const* d_in, const int* in_sizes, int n_in,
                              void* d_out, int out_size, void* d_ws, size_t ws_size,
                              hipStream_t stream) {
  (void)in_sizes; (void)n_in; (void)out_size; (void)ws_size;
  const float* hidden_c   = (const float*)d_in[0];
  const float* hidden_h   = (const float*)d_in[1];
  const float* state_emb  = (const float*)d_in[2];
  const unsigned char* dones = (const unsigned char*)d_in[3];  // bool array
  const float* action_emb = (const float*)d_in[4];
  const float* W1  = (const float*)d_in[5];
  const float* b1  = (const float*)d_in[6];
  const float* Wi  = (const float*)d_in[7];
  const float* Wh  = (const float*)d_in[8];
  const float* b_lstm = (const float*)d_in[9];
  const float* Wp  = (const float*)d_in[10];
  const float* bp  = (const float*)d_in[11];
  const float* Wa  = (const float*)d_in[12];
  const float* ba  = (const float*)d_in[13];
  const float* Wv  = (const float*)d_in[14];
  const float* bv  = (const float*)d_in[15];
  float* out = (float*)d_out;

  // ---- carve workspace ----
  char* ws = (char*)d_ws;
  auto carve = [&](size_t bytes) {
    char* p = ws;
    ws += (bytes + 255) & ~(size_t)255;
    return p;
  };
  __bf16* xin_bf = (__bf16*)carve((size_t)ROWS * D_STATE * 2);  // state_emb bf16
  __bf16* W1p    = (__bf16*)carve((size_t)D_STATE * H_DIM * 2); // packed
  __bf16* Wip    = (__bf16*)carve((size_t)H_DIM * G_DIM * 2);   // packed
  __bf16* Whp    = (__bf16*)carve((size_t)H_DIM * G_DIM * 2);   // packed
  __bf16* Wpp    = (__bf16*)carve((size_t)H_DIM * D_BERT * 2);  // packed
  __bf16* xbf    = (__bf16*)carve((size_t)ROWS * H_DIM * 2);    // relu(x@W1+b1)
  float*  Gx     = (float*) carve((size_t)ROWS * G_DIM * 4);    // x@Wi + b_lstm
  __bf16* ysbf   = (__bf16*)carve((size_t)ROWS * H_DIM * 2);
  float*  proj   = (float*) carve((size_t)ROWS * D_BERT * 4);
  float*  vbuf   = (float*) carve((size_t)ROWS * 4);
  float*  awbuf  = (float*) carve((size_t)A_DIM * D_BERT * 4);
  __bf16* hp0    = (__bf16*)carve((size_t)B_DIM * H_DIM * 2);   // packed A h ping
  __bf16* hp1    = (__bf16*)carve((size_t)B_DIM * H_DIM * 2);   // packed A h pong
  unsigned int* counter = (unsigned int*)carve(256);

  // ---- 1) weight pre-swizzle into fragment-major bf16 ----
  auto packB = [&](const float* src, __bf16* dst, int K, int N) {
    pack_b_kernel<<<(K * N + 255) / 256, 256, 0, stream>>>(src, dst, K, N);
  };
  packB(W1, W1p, D_STATE, H_DIM);
  packB(Wi, Wip, H_DIM, G_DIM);
  packB(Wh, Whp, H_DIM, G_DIM);
  packB(Wp, Wpp, H_DIM, D_BERT);
  cvt_f32_bf16_kernel<<<(ROWS * D_STATE + 255) / 256, 256, 0, stream>>>(
      state_emb, xin_bf, ROWS * D_STATE);

  // ---- 2) init LSTM hidden state (packed bf16) + barrier counter ----
  init_h_kernel<<<(B_DIM * H_DIM + 255) / 256, 256, 0, stream>>>(
      hidden_h, hp0, counter, B_DIM * H_DIM);

  // ---- 3) x = relu(state_emb @ W1 + b1) -> bf16 ----
  gemm_wmma_kernel<true, true><<<dim3(H_DIM / 64, ROWS / 64), 128, 0, stream>>>(
      xin_bf, W1p, b1, xbf, D_STATE, H_DIM);

  // ---- 4) Gx = x @ Wi + b_lstm (hoisted out of the scan) -> f32 ----
  gemm_wmma_kernel<false, false><<<dim3(G_DIM / 64, ROWS / 64), 128, 0, stream>>>(
      xbf, Wip, b_lstm, Gx, H_DIM, G_DIM);

  // ---- 5) persistent LSTM scan: ONE launch for all 128 steps ----
  lstm_scan_kernel<<<SCAN_BLOCKS, 128, 0, stream>>>(
      Whp, Gx, dones, hidden_c, hp0, hp1, ysbf, out, counter);

  // ---- 6) proj = ys @ Wp + bp -> f32 ----
  gemm_wmma_kernel<false, false><<<dim3(D_BERT / 64, ROWS / 64), 128, 0, stream>>>(
      ysbf, Wpp, bp, proj, H_DIM, D_BERT);

  // ---- 7) v = ys @ Wv + bv ----
  v_kernel<<<(ROWS + 255) / 256, 256, 0, stream>>>(ysbf, Wv, bv, vbuf);

  // ---- 8) q = v + ba + proj . (action_emb * Wa) ----
  aw_kernel<<<(A_DIM * D_BERT + 255) / 256, 256, 0, stream>>>(action_emb, Wa, awbuf);
  q_kernel<<<(ROWS * A_DIM + 255) / 256, 256, 0, stream>>>(
      proj, awbuf, ba, vbuf, out + 2 * B_DIM * H_DIM);
}
